// DualBiGRU_76235669504142
// MI455X (gfx1250) — compile-verified, matrix operand order checked by reading, not run on hardware
//
#include <hip/hip_runtime.h>

typedef __bf16 bf16_t;
typedef __attribute__((ext_vector_type(16))) __bf16 v16bf;
typedef __attribute__((ext_vector_type(8)))  float  v8f;

#define WMMA_BF16(a, b, c) \
  __builtin_amdgcn_wmma_f32_16x16x32_bf16(false, (a), false, (b), (short)0, (c), false, false)

// fast math: hardware v_rcp_f32 / v_exp_f32, no IEEE div chain
__device__ __forceinline__ float fast_rcp(float x) { return __builtin_amdgcn_rcpf(x); }
__device__ __forceinline__ float sigmoidf_(float x) { return fast_rcp(1.0f + __expf(-x)); }
__device__ __forceinline__ float tanhf_(float x)    { return 2.0f * fast_rcp(1.0f + __expf(-2.0f * x)) - 1.0f; }

// A-fragment staging layout in LDS: [mblock][kb][lane(32)][i(16)] bf16, so each
// lane's 16 halves of a 16x32 bf16 A fragment are contiguous (2x ds_load_b128).
// K mapping inside a 32-wide k-block follows the ISA 16-bit A layout:
//   lane<16 : i<8 -> K=i,     i>=8 -> K=16+(i-8)
//   lane>=16: i<8 -> K=8+i,   i>=8 -> K=24+(i-8)
__device__ __forceinline__ int stage_idx(int mw, int k, int mloc, int ktiles) {
  int kb = k >> 5, ko = k & 31;
  int g = (ko >> 3) & 1;
  int i = (ko & 7) + ((ko & 16) ? 8 : 0);
  return ((mw * ktiles + kb) * 32 + g * 16 + (mloc & 15)) * 16 + i;
}

// Pack f32 row-major W[K x N] into bf16 WMMA B fragments (32x16 tiles),
// tile index = kb*(N/16)+nb, element order = lane*16+i, mirrored K mapping.
extern "C" __global__ void pack_weight_kernel(const float* __restrict__ src,
                                              bf16_t* __restrict__ dst,
                                              int N, int Kvalid, int Ktiles) {
  int idx = blockIdx.x * blockDim.x + threadIdx.x;
  int total = Ktiles * (N >> 4) * 512;
  if (idx >= total) return;
  int tile = idx >> 9;
  int e = idx & 511;
  int lane = e >> 4, i = e & 15;
  int ntiles = N >> 4;
  int kb = tile / ntiles, nb = tile - kb * ntiles;
  int g = lane >> 4;
  int k = kb * 32 + ((i < 8) ? (g * 8 + i) : (16 + g * 8 + (i - 8)));
  int n = nb * 16 + (lane & 15);
  float v = (k < Kvalid) ? src[(size_t)k * N + n] : 0.0f;
  dst[idx] = (bf16_t)v;
}

// ---------------- Persistent GRU kernel ----------------
// Workgroup: 256 threads = 8 waves, 128 batch rows (one 16-row m-block/wave).
// B (weights) staged global->LDS in double-buffered 2-nblock chunks and shared
// by all 8 waves: 8x less L2 traffic than per-wave B streaming.
#define GRU_H   256
#define GRU_X   64
#define GRU_KB  10     // K=320 -> 10 k-blocks of 32
#define GRU_T   256
#define GRU_MT  128    // batch rows per workgroup (8 waves x 16 rows)

// dynamic LDS layout (bytes):
//  stage : 8*10*32*16*2 = 81920   (bf16 A fragments, [x|h])
//  bbuf  : 2*20*512*2   = 40960   (double-buffered B chunk: 20 tiles of 32x16)
//  ubuf  : 128*256*2    = 65536   (bf16 u gate)
//  hbuf  : 128*256*4    = 131072  (f32 hidden state)
//  gbias : 512*4        = 2048    (f32 gate bias)
//  cbias : 256*4        = 1024    (f32 candidate bias)
#define GRU_LDS (81920 + 40960 + 65536 + 131072 + 2048 + 1024)

extern "C" __global__ __launch_bounds__(256)
void gru_kernel(const float* __restrict__ seq,
                const bf16_t* __restrict__ pkG_fw, const bf16_t* __restrict__ pkC_fw,
                const bf16_t* __restrict__ pkG_bw, const bf16_t* __restrict__ pkC_bw,
                const float* __restrict__ gb_fw, const float* __restrict__ cb_fw,
                const float* __restrict__ gb_bw, const float* __restrict__ cb_bw,
                bf16_t* __restrict__ hcat) {
  extern __shared__ __align__(32) char smem[];
  bf16_t* stage = (bf16_t*)smem;
  bf16_t* bbuf  = (bf16_t*)(smem + 81920);
  bf16_t* ubuf  = (bf16_t*)(smem + 81920 + 40960);
  float*  hbuf  = (float*)(smem + 81920 + 40960 + 65536);
  float*  gb_s  = (float*)(smem + 81920 + 40960 + 65536 + 131072);
  float*  cb_s  = (float*)(smem + 81920 + 40960 + 65536 + 131072 + 2048);

  const int tid = threadIdx.x;
  const int run = blockIdx.x >> 3;            // 0=tgt_fw 1=tgt_bw 2=prb_fw 3=prb_bw
  const int mtile = blockIdx.x & 7;
  const int rowBase = mtile * GRU_MT;
  const bool bw = (run & 1);
  const int tBase = (run >= 2) ? 256 : 0;
  const bf16_t* pkG = bw ? pkG_bw : pkG_fw;
  const bf16_t* pkC = bw ? pkC_bw : pkC_fw;
  const float* gb = bw ? gb_bw : gb_fw;
  const float* cb = bw ? cb_bw : cb_fw;

  // zero hidden state + staging (h region must start at 0); stage biases in LDS
  for (int e = tid; e < GRU_MT * GRU_H; e += 256) hbuf[e] = 0.0f;
  {
    unsigned int* sz = (unsigned int*)stage;
    for (int e = tid; e < 20480; e += 256) sz[e] = 0u;
  }
  if (tid < 512) gb_s[tid] = gb[tid];
  if (tid < 256) cb_s[tid] = cb[tid];

  const int wave = tid >> 5;
  const int lane = tid & 31;
  const int lgrp = lane >> 4;
  const int lmod = lane & 15;
  const v16bf* stv = (const v16bf*)stage;
  const v16bf* pgv = (const v16bf*)pkG;
  const v16bf* pcv = (const v16bf*)pkC;
  v16bf* bbv = (v16bf*)bbuf;

  for (int t = 0; t < GRU_T; ++t) {
    __syncthreads();
    const int tIdx = tBase + (bw ? (GRU_T - 1 - t) : t);
    // cooperatively load x_t (128 rows x 64 cols f32) into bf16 staging (k<64)
    for (int e = tid; e < GRU_MT * GRU_X; e += 256) {
      int mloc = e >> 6, k = e & 63;
      float xv = seq[((size_t)(rowBase + mloc) * 512 + tIdx) * 64 + k];
      stage[stage_idx(mloc >> 4, k, mloc, GRU_KB)] = (bf16_t)xv;
    }
    // prefetch next timestep's x rows into cache
    if (t + 1 < GRU_T) {
      int tn = tBase + (bw ? (GRU_T - 2 - t) : (t + 1));
      int pr = tid >> 1, ph = (tid & 1) * 32;
      __builtin_prefetch(&seq[((size_t)(rowBase + pr) * 512 + tn) * 64 + ph], 0, 1);
    }
    __syncthreads();

    // A fragments for [x | h] (each wave: its own 16-row block)
    v16bf af[GRU_KB];
#pragma unroll
    for (int kb = 0; kb < GRU_KB; ++kb)
      af[kb] = stv[(wave * GRU_KB + kb) * 32 + lane];

    // ---- GEMM1: gates (32 n-blocks; nb<16 -> r, nb>=16 -> u) ----
    // preload chunk 0 (nb 0,1) into bbuf half 0
    for (int e = tid; e < 640; e += 256) {
      int tl = e >> 5, ln = e & 31;
      int kb = tl >> 1, nbo = tl & 1;
      bbv[tl * 32 + ln] = pgv[(kb * 32 + nbo) * 32 + ln];
    }
    __syncthreads();
    for (int c = 0; c < 16; ++c) {
      const int cur = c & 1;
      if (c < 15) {
        int nb0 = 2 * (c + 1);
        for (int e = tid; e < 640; e += 256) {
          int tl = e >> 5, ln = e & 31;
          int kb = tl >> 1, nbo = tl & 1;
          bbv[((cur ^ 1) * 20 + tl) * 32 + ln] = pgv[(kb * 32 + nb0 + nbo) * 32 + ln];
        }
      }
      for (int nbo = 0; nbo < 2; ++nbo) {
        const int nb = 2 * c + nbo;
        v8f acc = {};
#pragma unroll
        for (int kb = 0; kb < GRU_KB; ++kb) {
          v16bf bfr = bbv[(cur * 20 + kb * 2 + nbo) * 32 + lane];
          acc = WMMA_BF16(af[kb], bfr, acc);
        }
        if (nb < 16) {
          // r gate -> r*h into bf16 A staging (candidate operand)
#pragma unroll
          for (int v = 0; v < 8; ++v) {
            int mloc = wave * 16 + lgrp * 8 + v;
            int n = nb * 16 + lmod;
            float r = sigmoidf_(acc[v] + gb_s[n]);
            float rh = r * hbuf[mloc * GRU_H + n];
            stage[stage_idx(wave, 64 + n, mloc, GRU_KB)] = (bf16_t)rh;
          }
        } else {
          // u gate -> bf16 LDS
#pragma unroll
          for (int v = 0; v < 8; ++v) {
            int mloc = wave * 16 + lgrp * 8 + v;
            int n = (nb - 16) * 16 + lmod;
            float u = sigmoidf_(acc[v] + gb_s[256 + n]);
            ubuf[mloc * GRU_H + n] = (bf16_t)u;
          }
        }
      }
      __syncthreads();
    }

    // reload h-part A frags: they now contain r*h (k-blocks 2..9)
#pragma unroll
    for (int kb = 2; kb < GRU_KB; ++kb)
      af[kb] = stv[(wave * GRU_KB + kb) * 32 + lane];

    // ---- GEMM2: candidate (16 n-blocks) + combine ----
    for (int e = tid; e < 640; e += 256) {
      int tl = e >> 5, ln = e & 31;
      int kb = tl >> 1, nbo = tl & 1;
      bbv[tl * 32 + ln] = pcv[(kb * 16 + nbo) * 32 + ln];
    }
    __syncthreads();
    for (int c = 0; c < 8; ++c) {
      const int cur = c & 1;
      if (c < 7) {
        int nb0 = 2 * (c + 1);
        for (int e = tid; e < 640; e += 256) {
          int tl = e >> 5, ln = e & 31;
          int kb = tl >> 1, nbo = tl & 1;
          bbv[((cur ^ 1) * 20 + tl) * 32 + ln] = pcv[(kb * 16 + nb0 + nbo) * 32 + ln];
        }
      }
      for (int nbo = 0; nbo < 2; ++nbo) {
        const int nb = 2 * c + nbo;
        v8f acc = {};
#pragma unroll
        for (int kb = 0; kb < GRU_KB; ++kb) {
          v16bf bfr = bbv[(cur * 20 + kb * 2 + nbo) * 32 + lane];
          acc = WMMA_BF16(af[kb], bfr, acc);
        }
#pragma unroll
        for (int v = 0; v < 8; ++v) {
          int mloc = wave * 16 + lgrp * 8 + v;
          int n = nb * 16 + lmod;
          float cc = tanhf_(acc[v] + cb_s[n]);
          float u = (float)ubuf[mloc * GRU_H + n];
          float hn = u * hbuf[mloc * GRU_H + n] + (1.0f - u) * cc;
          hbuf[mloc * GRU_H + n] = hn;
          stage[stage_idx(wave, 64 + n, mloc, GRU_KB)] = (bf16_t)hn;
        }
      }
      __syncthreads();
    }
  }

  __syncthreads();
  // final hidden -> hcat[b][run*256 + n] (bf16, MLP A operand)
  for (int e = tid; e < GRU_MT * GRU_H; e += 256) {
    int mloc = e >> 8, n = e & 255;
    hcat[(size_t)(rowBase + mloc) * 1056 + run * 256 + n] = (bf16_t)hbuf[e];
  }
}

// ---------------- Generic WMMA GEMM + bias + relu (MLP layers) ----------------
extern "C" __global__ __launch_bounds__(128)
void mlp_gemm_kernel(const bf16_t* __restrict__ A, const bf16_t* __restrict__ pkW,
                     const float* __restrict__ bias, int N, int K, int Ktiles,
                     float* __restrict__ outF, bf16_t* __restrict__ outB) {
  extern __shared__ __align__(32) char smem[];
  bf16_t* stage = (bf16_t*)smem;                       // 4 * Ktiles * 512 bf16
  float*  bias_s = (float*)(smem + (size_t)Ktiles * 4096);
  const int tid = threadIdx.x;
  const int rowBase = blockIdx.x * 64;
  for (int e = tid; e < 64 * K; e += 128) {
    int mloc = e / K, k = e - mloc * K;
    stage[stage_idx(mloc >> 4, k, mloc, Ktiles)] = A[(size_t)(rowBase + mloc) * K + k];
  }
  for (int e = tid; e < N; e += 128) bias_s[e] = bias[e];
  __syncthreads();
  const int wave = tid >> 5, lane = tid & 31;
  const int lgrp = lane >> 4, lmod = lane & 15;
  const v16bf* stv = (const v16bf*)stage;
  const v16bf* pwv = (const v16bf*)pkW;
  const int ntiles = N >> 4;
  for (int nb = 0; nb < ntiles; ++nb) {
    v8f acc = {};
    for (int kb = 0; kb < Ktiles; ++kb) {
      v16bf afr = stv[(wave * Ktiles + kb) * 32 + lane];
      v16bf bfr = pwv[(kb * ntiles + nb) * 32 + lane];
      acc = WMMA_BF16(afr, bfr, acc);
    }
#pragma unroll
    for (int v = 0; v < 8; ++v) {
      int mloc = wave * 16 + lgrp * 8 + v;
      int n = nb * 16 + lmod;
      float val = fmaxf(acc[v] + bias_s[n], 0.0f);
      size_t o = (size_t)(rowBase + mloc) * N + n;
      if (outB) outB[o] = (bf16_t)val;
      else      outF[o] = val;
    }
  }
}

// fill hcat cols 1024..1055: global features (8) then zero pad (24)
extern "C" __global__ void hcat_tail_kernel(const float* __restrict__ gf,
                                            bf16_t* __restrict__ hcat) {
  int idx = blockIdx.x * blockDim.x + threadIdx.x;
  if (idx >= 1024 * 32) return;
  int b = idx >> 5, c = idx & 31;
  float v = (c < 8) ? gf[b * 8 + c] : 0.0f;
  hcat[(size_t)b * 1056 + 1024 + c] = (bf16_t)v;
}

// final head: out[b] = [h@w3+b3][0], |[h@w3+b3][1]|
extern "C" __global__ void head_kernel(const float* __restrict__ act2,
                                       const float* __restrict__ w3,
                                       const float* __restrict__ b3,
                                       float* __restrict__ out) {
  int b = blockIdx.x * blockDim.x + threadIdx.x;
  if (b >= 1024) return;
  float s0 = b3[0], s1 = b3[1];
  for (int k = 0; k < 256; ++k) {
    float a = act2[b * 256 + k];
    s0 += a * w3[k * 2 + 0];
    s1 += a * w3[k * 2 + 1];
  }
  out[b * 2 + 0] = s0;
  out[b * 2 + 1] = fabsf(s1);
}

extern "C" void kernel_launch(void* const* d_in, const int* in_sizes, int n_in,
                              void* d_out, int out_size, void* d_ws, size_t ws_size,
                              hipStream_t stream) {
  const float* seq   = (const float*)d_in[0];
  const float* gf    = (const float*)d_in[1];
  const float* fw_gw = (const float*)d_in[2];
  const float* fw_gb = (const float*)d_in[3];
  const float* fw_cw = (const float*)d_in[4];
  const float* fw_cb = (const float*)d_in[5];
  const float* bw_gw = (const float*)d_in[6];
  const float* bw_gb = (const float*)d_in[7];
  const float* bw_cw = (const float*)d_in[8];
  const float* bw_cb = (const float*)d_in[9];
  const float* w1 = (const float*)d_in[10];
  const float* b1 = (const float*)d_in[11];
  const float* w2 = (const float*)d_in[12];
  const float* b2 = (const float*)d_in[13];
  const float* w3 = (const float*)d_in[14];
  const float* b3 = (const float*)d_in[15];
  float* out = (float*)d_out;

  char* ws = (char*)d_ws;
  bf16_t* pkG_fw = (bf16_t*)(ws + 0);         // 320x512 bf16 = 327680 B
  bf16_t* pkC_fw = (bf16_t*)(ws + 327680);    // 320x256 bf16 = 163840 B
  bf16_t* pkG_bw = (bf16_t*)(ws + 491520);
  bf16_t* pkC_bw = (bf16_t*)(ws + 819200);
  bf16_t* pkW1   = (bf16_t*)(ws + 983040);    // 1056x512 bf16 = 1081344 B
  bf16_t* pkW2   = (bf16_t*)(ws + 2064384);   // 512x256 bf16  = 262144 B
  bf16_t* hcat   = (bf16_t*)(ws + 2326528);   // 1024x1056 bf16 = 2162688 B
  bf16_t* act1   = (bf16_t*)(ws + 4489216);   // 1024x512 bf16  = 1048576 B
  float*  act2   = (float*)(ws + 5537792);    // 1024x256 f32   = 1048576 B

  auto packLaunch = [&](const float* src, bf16_t* dst, int N, int Kvalid, int Ktiles) {
    int total = Ktiles * (N >> 4) * 512;
    pack_weight_kernel<<<(total + 255) / 256, 256, 0, stream>>>(src, dst, N, Kvalid, Ktiles);
  };
  packLaunch(fw_gw, pkG_fw, 512, 320, 10);
  packLaunch(fw_cw, pkC_fw, 256, 320, 10);
  packLaunch(bw_gw, pkG_bw, 512, 320, 10);
  packLaunch(bw_cw, pkC_bw, 256, 320, 10);
  packLaunch(w1,    pkW1,   512, 1032, 33);   // pad K 1032 -> 1056 with zeros
  packLaunch(w2,    pkW2,   256, 512,  16);

  hcat_tail_kernel<<<(1024 * 32 + 255) / 256, 256, 0, stream>>>(gf, hcat);

  // 4 runs x 8 batch tiles of 128 rows; 256 threads; ~315KB dynamic LDS
  gru_kernel<<<32, 256, GRU_LDS, stream>>>(seq, pkG_fw, pkC_fw, pkG_bw, pkC_bw,
                                           fw_gb, fw_cb, bw_gb, bw_cb, hcat);

  mlp_gemm_kernel<<<16, 128, 33 * 4096 + 2048, stream>>>(hcat, pkW1, b1, 512, 1056, 33,
                                                         nullptr, act1);
  mlp_gemm_kernel<<<16, 128, 16 * 4096 + 1024, stream>>>(act1, pkW2, b2, 256, 512, 16,
                                                         act2, nullptr);
  head_kernel<<<4, 256, 0, stream>>>(act2, w3, b3, out);
}
